// Qwen3MoeDecoderLayer_50268297232750
// MI455X (gfx1250) — matrix operand order repaired and checked by reading, not compiled, not run on hardware
//
#include <hip/hip_runtime.h>
#include <math.h>

// ---------------------------------------------------------------------------
// Qwen3-MoE decoder layer for gfx1250 (MI455X).
// Shapes: B=1, S=2048, H=2048, NH=32, NKV=4, HD=128, E=16, I=768, TOP_K=4.
// All GEMMs run v_wmma_f32_16x16x32_bf16 on pre-converted bf16 operands:
// the inner loop is global loads + wmma (16B vector loads except NN-form B). All 4 B fragments of a
// K-step are preloaded into distinct registers so the scheduler can stagger
// s_wait_loadcnt instead of draining to 0 before every WMMA.
// ---------------------------------------------------------------------------

typedef __attribute__((ext_vector_type(16))) __bf16 v16bf;
typedef __attribute__((ext_vector_type(8)))  __bf16 v8bf;
typedef __attribute__((ext_vector_type(8)))  float  v8f;

__device__ __forceinline__ __bf16 to_bf16(float f) {
    unsigned u = __builtin_bit_cast(unsigned, f);
    unsigned r = u + 0x7FFFu + ((u >> 16) & 1u);   // round-to-nearest-even
    unsigned short h = (unsigned short)(r >> 16);
    return __builtin_bit_cast(__bf16, h);
}

// ---------------------------------------------------------------------------
// Generic WMMA GEMM on bf16 operands:
//   C[M,N] = alpha * A[M,K] * B  (+ addv[M,N] if HAS_ADD)
//   B_IS_NT = true : B stored row-major [N,K]  (C = A @ B^T, weight form)
//   B_IS_NT = false: B stored row-major [K,N]  (C = A @ B)
//   OUT_BF16 selects bf16 vs fp32 C.
// Block = 256 threads = 8 waves arranged 4(M) x 2(N); wave strip 16x64
// (4 WMMAs per K-step sharing one A fragment). Block tile 64x128.
// Requires M % 64 == 0, N % 128 == 0, K % 32 == 0 (true at every call site),
// so EXEC is always all-ones as WMMA requires.
// ---------------------------------------------------------------------------
template <bool B_IS_NT, bool HAS_ADD, bool OUT_BF16>
__global__ void __launch_bounds__(256)
gemm_wmma_kernel(const __bf16* __restrict__ A, int lda,
                 const __bf16* __restrict__ B, int ldb,
                 void* __restrict__ Cv, int ldc,
                 int M, int N, int K, float alpha,
                 const float* __restrict__ addv, int ldadd)
{
    const int lane = threadIdx.x & 31;
    const int wave = threadIdx.x >> 5;        // 0..7
    const int m0   = blockIdx.y * 64 + (wave >> 1) * 16;   // 4 waves along M
    const int n0   = blockIdx.x * 128 + (wave & 1) * 64;   // 2 waves along N
    const int half = lane >> 4;               // 0 or 1
    const int l16  = lane & 15;

    const __bf16* Arow = A + (size_t)(m0 + l16) * lda;

    v8f acc[4] = {};
    for (int kb = 0; kb < K; kb += 32) {
        // A fragment (16x32 bf16, contiguous layout): lane half h covers
        // k = kb+h*16+[0..15]; two 16B vector loads.
        const __bf16* ap = Arow + kb + half * 16;
        __builtin_prefetch((const void*)(ap + 128), 0, 0);
        const v8bf alo = *(const v8bf*)(ap);
        const v8bf ahi = *(const v8bf*)(ap + 8);
        const v16bf a = __builtin_shufflevector(alo, ahi,
            0, 1, 2, 3, 4, 5, 6, 7, 8, 9, 10, 11, 12, 13, 14, 15);

        // Preload ALL 4 B fragments into distinct registers first, so the
        // consuming WMMAs below can use staggered s_wait_loadcnt values
        // instead of draining all loads before each matrix op.
        v16bf b[4];
        if constexpr (B_IS_NT) {
#pragma unroll
            for (int j = 0; j < 4; ++j) {
                const __bf16* bp = B + (size_t)(n0 + j * 16 + l16) * ldb
                                     + kb + half * 16;
                const v8bf blo = *(const v8bf*)(bp);
                const v8bf bhi = *(const v8bf*)(bp + 8);
                b[j] = __builtin_shufflevector(blo, bhi,
                    0, 1, 2, 3, 4, 5, 6, 7, 8, 9, 10, 11, 12, 13, 14, 15);
            }
        } else {
#pragma unroll
            for (int j = 0; j < 4; ++j) {
                const __bf16* bp = B + (size_t)(kb + half * 16) * ldb
                                     + (n0 + j * 16 + l16);
#pragma unroll
                for (int e = 0; e < 16; ++e) b[j][e] = bp[(size_t)e * ldb];
            }
        }

        // 4 back-to-back WMMAs sharing the A fragment.
#pragma unroll
        for (int j = 0; j < 4; ++j)
            acc[j] = __builtin_amdgcn_wmma_f32_16x16x32_bf16(
                false, a, false, b[j], (short)0, acc[j], false, false);
    }

    // Epilogue: C/D layout -> VGPR r holds row m0 + r + 8*half, col n.
    const size_t rowbase = (size_t)(m0 + 8 * half);
#pragma unroll
    for (int j = 0; j < 4; ++j) {
        const int n = n0 + j * 16 + l16;
        if constexpr (OUT_BF16) {
            __bf16* cp = (__bf16*)Cv + rowbase * ldc + n;
#pragma unroll
            for (int r = 0; r < 8; ++r)
                cp[(size_t)r * ldc] = to_bf16(alpha * acc[j][r]);
        } else if constexpr (HAS_ADD) {
            float* cp = (float*)Cv + rowbase * ldc + n;
            const float* addp = addv + rowbase * ldadd + n;
#pragma unroll
            for (int r = 0; r < 8; ++r)
                cp[(size_t)r * ldc] = alpha * acc[j][r] + addp[(size_t)r * ldadd];
        } else {
            float* cp = (float*)Cv + rowbase * ldc + n;
#pragma unroll
            for (int r = 0; r < 8; ++r)
                cp[(size_t)r * ldc] = alpha * acc[j][r];
        }
    }
}

// ---------------------------------------------------------------------------
// fp32 -> bf16 streaming conversion, 4 elements/thread (n % 4 == 0)
// ---------------------------------------------------------------------------
__global__ void __launch_bounds__(256)
cvt_bf16_kernel(const float* __restrict__ in, __bf16* __restrict__ out, long n4)
{
    const long i = ((long)blockIdx.x * 256 + threadIdx.x) * 4;
    if (i < n4 * 4) {
        const float4 v = *(const float4*)(in + i);
        __bf16* o = out + i;
        o[0] = to_bf16(v.x); o[1] = to_bf16(v.y);
        o[2] = to_bf16(v.z); o[3] = to_bf16(v.w);
    }
}

// ---------------------------------------------------------------------------
// RMSNorm over rows of length H; writes bf16 (always) and fp32 (optional).
// Block = 256 threads = 8 waves per row.
// ---------------------------------------------------------------------------
__global__ void __launch_bounds__(256)
rmsnorm_kernel(const float* __restrict__ x, const float* __restrict__ w,
               __bf16* __restrict__ yb, float* __restrict__ yf, int H)
{
    const int row = blockIdx.x;
    const float* xr = x + (size_t)row * H;
    float ss = 0.f;
    for (int j = threadIdx.x; j < H; j += 256) { float v = xr[j]; ss += v * v; }
#pragma unroll
    for (int m = 16; m > 0; m >>= 1) ss += __shfl_xor(ss, m, 32);
    __shared__ float red[8];
    const int wv = threadIdx.x >> 5, ln = threadIdx.x & 31;
    if (ln == 0) red[wv] = ss;
    __syncthreads();
    float tot = 0.f;
#pragma unroll
    for (int i = 0; i < 8; ++i) tot += red[i];
    const float scale = rsqrtf(tot / (float)H + 1e-6f);
    __bf16* ybr = yb + (size_t)row * H;
    for (int j = threadIdx.x; j < H; j += 256) {
        const float v = w[j] * xr[j] * scale;
        ybr[j] = to_bf16(v);
        if (yf) yf[(size_t)row * H + j] = v;
    }
}

// ---------------------------------------------------------------------------
// Fused per-(token,head) q/k RMSNorm (over HD=128) + RoPE.
// Reads fp32 projection output, writes bf16. One 128-thread block per row.
// pe: [2, S, 128] (cos then sin).
// ---------------------------------------------------------------------------
__global__ void __launch_bounds__(128)
qknorm_rope_kernel(const float* __restrict__ qk, __bf16* __restrict__ outb,
                   const float* __restrict__ nw, const float* __restrict__ pe,
                   int S, int nh)
{
    const int blk = blockIdx.x;
    const int s = blk / nh;
    const int h = blk - s * nh;
    const int d = threadIdx.x;                  // 0..127
    const size_t off = ((size_t)s * nh + h) * 128;
    const float v = qk[off + d];

    __shared__ float sh[128];
    __shared__ float smean;
    sh[d] = v * v;
    __syncthreads();
    for (int o = 64; o > 0; o >>= 1) {
        if (d < o) sh[d] += sh[d + o];
        __syncthreads();
    }
    if (d == 0) smean = sh[0] * (1.f / 128.f);
    __syncthreads();

    const float normed = v * rsqrtf(smean + 1e-6f) * nw[d];
    sh[d] = normed;
    __syncthreads();
    const float rh = (d < 64) ? -sh[d + 64] : sh[d - 64];
    const float c  = pe[(size_t)s * 128 + d];
    const float si = pe[(size_t)(S + s) * 128 + d];
    outb[off + d] = to_bf16(normed * c + rh * si);
}

// ---------------------------------------------------------------------------
// Row softmax over fp32 [rows, N]; writes normalized probs as bf16.
// Block = 256 threads per row.
// ---------------------------------------------------------------------------
__global__ void __launch_bounds__(256)
softmax_rows_kernel(float* __restrict__ s, __bf16* __restrict__ pb, int N)
{
    const int row = blockIdx.x;
    float* r = s + (size_t)row * N;
    __shared__ float red[8];
    const int wv = threadIdx.x >> 5, ln = threadIdx.x & 31;

    float mx = -3.0e38f;
    for (int j = threadIdx.x; j < N; j += 256) mx = fmaxf(mx, r[j]);
#pragma unroll
    for (int m = 16; m > 0; m >>= 1) mx = fmaxf(mx, __shfl_xor(mx, m, 32));
    if (ln == 0) red[wv] = mx;
    __syncthreads();
    float gmx = red[0];
#pragma unroll
    for (int i = 1; i < 8; ++i) gmx = fmaxf(gmx, red[i]);
    __syncthreads();

    float sum = 0.f;
    for (int j = threadIdx.x; j < N; j += 256) {
        const float e = expf(r[j] - gmx);
        r[j] = e;
        sum += e;
    }
#pragma unroll
    for (int m = 16; m > 0; m >>= 1) sum += __shfl_xor(sum, m, 32);
    if (ln == 0) red[wv] = sum;
    __syncthreads();
    float gs = 0.f;
#pragma unroll
    for (int i = 0; i < 8; ++i) gs += red[i];
    const float inv = 1.f / gs;
    __bf16* pr = pb + (size_t)row * N;
    for (int j = threadIdx.x; j < N; j += 256) pr[j] = to_bf16(r[j] * inv);
}

// ---------------------------------------------------------------------------
// Router logits: one wave per token, 16 experts, K = H (fp32 path so expert
// selection matches the fp32 reference).
// ---------------------------------------------------------------------------
__global__ void __launch_bounds__(32)
gate_logits_kernel(const float* __restrict__ h2, const float* __restrict__ gw,
                   float* __restrict__ logits, int H)
{
    const int t = blockIdx.x;
    const int lane = threadIdx.x;
    const float* hr = h2 + (size_t)t * H;
    float acc[16];
#pragma unroll
    for (int e = 0; e < 16; ++e) acc[e] = 0.f;
    for (int k = lane; k < H; k += 32) {
        const float hv = hr[k];
#pragma unroll
        for (int e = 0; e < 16; ++e) acc[e] += hv * gw[(size_t)e * H + k];
    }
#pragma unroll
    for (int e = 0; e < 16; ++e) {
        float v = acc[e];
#pragma unroll
        for (int m = 16; m > 0; m >>= 1) v += __shfl_xor(v, m, 32);
        if (lane == 0) logits[(size_t)t * 16 + e] = v;
    }
}

// ---------------------------------------------------------------------------
// Softmax over 16 experts + top-4 + renorm -> dense combine weights [T,16]
// ---------------------------------------------------------------------------
__global__ void __launch_bounds__(256)
routing_kernel(const float* __restrict__ logits, float* __restrict__ comb, int T)
{
    const int t = blockIdx.x * 256 + threadIdx.x;
    if (t >= T) return;
    float p[16];
    float mx = -3.0e38f;
#pragma unroll
    for (int e = 0; e < 16; ++e) { p[e] = logits[(size_t)t * 16 + e]; mx = fmaxf(mx, p[e]); }
    float sum = 0.f;
#pragma unroll
    for (int e = 0; e < 16; ++e) { p[e] = expf(p[e] - mx); sum += p[e]; }
    const float inv = 1.f / sum;
#pragma unroll
    for (int e = 0; e < 16; ++e) p[e] *= inv;

    unsigned taken = 0;
    int   sel[4];
    float wv[4];
    float wsum = 0.f;
#pragma unroll
    for (int k = 0; k < 4; ++k) {
        int best = 0;
        float bv = -1.f;
#pragma unroll
        for (int e = 0; e < 16; ++e) {
            const bool used = (taken >> e) & 1u;
            if (!used && p[e] > bv) { bv = p[e]; best = e; }
        }
        taken |= 1u << best;
        sel[k] = best;
        wv[k] = bv;
        wsum += bv;
    }
    float o[16];
#pragma unroll
    for (int e = 0; e < 16; ++e) o[e] = 0.f;
    const float winv = 1.f / wsum;
#pragma unroll
    for (int k = 0; k < 4; ++k) o[sel[k]] += wv[k] * winv;
#pragma unroll
    for (int e = 0; e < 16; ++e) comb[(size_t)t * 16 + e] = o[e];
}

// ---------------------------------------------------------------------------
// a = silu(g) * u  (fp32 in, bf16 out -> feeds the down-proj GEMM)
// ---------------------------------------------------------------------------
__global__ void __launch_bounds__(256)
silu_mul_kernel(const float* __restrict__ g, const float* __restrict__ u,
                __bf16* __restrict__ a, long n)
{
    const long i = (long)blockIdx.x * 256 + threadIdx.x;
    if (i < n) {
        const float gv = g[i];
        const float s = gv / (1.f + expf(-gv));
        a[i] = to_bf16(s * u[i]);
    }
}

// ---------------------------------------------------------------------------
// out[t,h] += comb[t,e] * y[t,h]
// ---------------------------------------------------------------------------
__global__ void __launch_bounds__(256)
moe_accum_kernel(const float* __restrict__ y, const float* __restrict__ comb,
                 int e, float* __restrict__ out, long n, int H, int E)
{
    const long i = (long)blockIdx.x * 256 + threadIdx.x;
    if (i < n) {
        const long t = i / H;
        out[i] += comb[(size_t)t * E + e] * y[i];
    }
}

// ---------------------------------------------------------------------------
// Host-side launch helpers
// ---------------------------------------------------------------------------
static inline void cvt_bf16(const float* in, __bf16* out, long n, hipStream_t st)
{
    const long n4 = n / 4;
    cvt_bf16_kernel<<<(unsigned)((n4 + 255) / 256), 256, 0, st>>>(in, out, n4);
}

static inline void gemm_nt_f32(const __bf16* A, int lda, const __bf16* B, int ldb,
                               float* C, int ldc, int M, int N, int K, float alpha,
                               const float* addv, int ldadd, hipStream_t st)
{
    dim3 grid((N + 127) / 128, (M + 63) / 64), blk(256);
    if (addv)
        gemm_wmma_kernel<true, true, false><<<grid, blk, 0, st>>>(
            A, lda, B, ldb, C, ldc, M, N, K, alpha, addv, ldadd);
    else
        gemm_wmma_kernel<true, false, false><<<grid, blk, 0, st>>>(
            A, lda, B, ldb, C, ldc, M, N, K, alpha, nullptr, 0);
}

static inline void gemm_nt_bf16(const __bf16* A, int lda, const __bf16* B, int ldb,
                                __bf16* C, int ldc, int M, int N, int K,
                                hipStream_t st)
{
    dim3 grid((N + 127) / 128, (M + 63) / 64), blk(256);
    gemm_wmma_kernel<true, false, true><<<grid, blk, 0, st>>>(
        A, lda, B, ldb, C, ldc, M, N, K, 1.f, nullptr, 0);
}

static inline void gemm_nn_bf16(const __bf16* A, int lda, const __bf16* B, int ldb,
                                __bf16* C, int ldc, int M, int N, int K,
                                hipStream_t st)
{
    dim3 grid((N + 127) / 128, (M + 63) / 64), blk(256);
    gemm_wmma_kernel<false, false, true><<<grid, blk, 0, st>>>(
        A, lda, B, ldb, C, ldc, M, N, K, 1.f, nullptr, 0);
}

extern "C" void kernel_launch(void* const* d_in, const int* in_sizes, int n_in,
                              void* d_out, int out_size, void* d_ws, size_t ws_size,
                              hipStream_t stream)
{
    (void)in_sizes; (void)n_in; (void)out_size; (void)ws_size;
    enum { S = 2048, H = 2048, NH = 32, NKV = 4, HD = 128, E = 16, I = 768,
           TOK = S, DQ = NH * HD, DKV = NKV * HD };

    const float* hidden   = (const float*)d_in[0];   // [1,S,H]
    const float* pe       = (const float*)d_in[1];   // [2,S,HD]
    const float* mask     = (const float*)d_in[2];   // [1,1,S,S]
    const float* q_w      = (const float*)d_in[3];   // [DQ,H]
    const float* k_w      = (const float*)d_in[4];   // [DKV,H]
    const float* v_w      = (const float*)d_in[5];   // [DKV,H]
    const float* o_w      = (const float*)d_in[6];   // [H,DQ]
    const float* q_norm_w = (const float*)d_in[7];   // [HD]
    const float* k_norm_w = (const float*)d_in[8];   // [HD]
    const float* ln1_w    = (const float*)d_in[9];   // [H]
    const float* ln2_w    = (const float*)d_in[10];  // [H]
    const float* gate_w   = (const float*)d_in[11];  // [E,H]
    const float* gp_w     = (const float*)d_in[12];  // [E,I,H]
    const float* up_w     = (const float*)d_in[13];  // [E,I,H]
    const float* dn_w     = (const float*)d_in[14];  // [E,H,I]
    float* out = (float*)d_out;                      // [1,S,H]

    // ---- workspace carve-up: fp32 region then bf16 region (~220 MB) ----
    float* wsf    = (float*)d_ws;
    float* q      = wsf;                        // [TOK,DQ]  fp32 (pre-RoPE)
    float* k      = q      + (size_t)TOK * DQ;  // [TOK,DKV] fp32 (pre-RoPE)
    float* scores = k      + (size_t)TOK * DKV; // [S,S]     fp32
    float* res    = scores + (size_t)S * S;     // [TOK,H]
    float* h2     = res    + (size_t)TOK * H;   // [TOK,H]
    float* logits = h2     + (size_t)TOK * H;   // [TOK,E]
    float* comb   = logits + (size_t)TOK * E;   // [TOK,E]
    float* gbuf   = comb   + (size_t)TOK * E;   // [TOK,I]
    float* ubuf   = gbuf   + (size_t)TOK * I;   // [TOK,I]
    float* ybuf   = ubuf   + (size_t)TOK * I;   // [TOK,H]

    __bf16* wsb  = (__bf16*)(ybuf + (size_t)TOK * H);
    __bf16* xnb  = wsb;                          // [TOK,H]   ln1 output
    __bf16* qb   = xnb  + (size_t)TOK * H;       // [TOK,DQ]  post-RoPE Q
    __bf16* kb   = qb   + (size_t)TOK * DQ;      // [TOK,DKV] post-RoPE K
    __bf16* vb   = kb   + (size_t)TOK * DKV;     // [TOK,DKV] V
    __bf16* pb   = vb   + (size_t)TOK * DKV;     // [S,S]     softmax probs
    __bf16* aob  = pb   + (size_t)S * S;         // [TOK,DQ]  attn out
    __bf16* h2b  = aob  + (size_t)TOK * DQ;      // [TOK,H]   ln2 output
    __bf16* abuf = h2b  + (size_t)TOK * H;       // [TOK,I]   silu(g)*u
    __bf16* qwb  = abuf + (size_t)TOK * I;       // [DQ,H]
    __bf16* kwb  = qwb  + (size_t)DQ * H;        // [DKV,H]
    __bf16* vwb  = kwb  + (size_t)DKV * H;       // [DKV,H]
    __bf16* owb  = vwb  + (size_t)DKV * H;       // [H,DQ]
    __bf16* wgb  = owb  + (size_t)H * DQ;        // [I,H] per-expert gate
    __bf16* wub  = wgb  + (size_t)I * H;         // [I,H] per-expert up
    __bf16* wdb  = wub  + (size_t)I * H;         // [H,I] per-expert down

    // ---- one-time weight conversions (each element converted exactly once)
    cvt_bf16(q_w, qwb, (long)DQ * H, stream);
    cvt_bf16(k_w, kwb, (long)DKV * H, stream);
    cvt_bf16(v_w, vwb, (long)DKV * H, stream);
    cvt_bf16(o_w, owb, (long)H * DQ, stream);

    // ---- attention block ----
    rmsnorm_kernel<<<TOK, 256, 0, stream>>>(hidden, ln1_w, xnb, nullptr, H);

    gemm_nt_f32(xnb, H, qwb, H, q, DQ,  TOK, DQ,  H, 1.f, nullptr, 0, stream);
    gemm_nt_f32(xnb, H, kwb, H, k, DKV, TOK, DKV, H, 1.f, nullptr, 0, stream);
    gemm_nt_bf16(xnb, H, vwb, H, vb, DKV, TOK, DKV, H, stream);

    qknorm_rope_kernel<<<S * NH,  128, 0, stream>>>(q, qb, q_norm_w, pe, S, NH);
    qknorm_rope_kernel<<<S * NKV, 128, 0, stream>>>(k, kb, k_norm_w, pe, S, NKV);

    const float scale = 0.08838834764831845f;  // 1/sqrt(128)
    for (int h = 0; h < NH; ++h) {
        const int kv = h / (NH / NKV);
        // scores = scale * Q_h K_h^T + mask
        gemm_nt_f32(qb + (size_t)h * HD, DQ, kb + (size_t)kv * HD, DKV,
                    scores, S, S, S, HD, scale, mask, S, stream);
        softmax_rows_kernel<<<S, 256, 0, stream>>>(scores, pb, S);
        // ao_h = P @ V_h
        gemm_nn_bf16(pb, S, vb + (size_t)kv * HD, DKV,
                     aob + (size_t)h * HD, DQ, S, HD, S, stream);
    }

    // res = hidden + ao @ o_w^T
    gemm_nt_f32(aob, DQ, owb, DQ, res, H, TOK, H, DQ, 1.f, hidden, H, stream);

    // ---- sparse MoE block ----
    rmsnorm_kernel<<<TOK, 256, 0, stream>>>(res, ln2_w, h2b, h2, H);
    gate_logits_kernel<<<TOK, 32, 0, stream>>>(h2, gate_w, logits, H);
    routing_kernel<<<(TOK + 255) / 256, 256, 0, stream>>>(logits, comb, TOK);

    // out starts as the attention residual
    hipMemcpyAsync(out, res, (size_t)TOK * H * sizeof(float),
                   hipMemcpyDeviceToDevice, stream);

    const long nTI = (long)TOK * I;
    const long nTH = (long)TOK * H;
    for (int e = 0; e < E; ++e) {
        cvt_bf16(gp_w + (size_t)e * I * H, wgb, (long)I * H, stream);
        cvt_bf16(up_w + (size_t)e * I * H, wub, (long)I * H, stream);
        cvt_bf16(dn_w + (size_t)e * H * I, wdb, (long)H * I, stream);
        gemm_nt_f32(h2b, H, wgb, H, gbuf, I, TOK, I, H, 1.f, nullptr, 0, stream);
        gemm_nt_f32(h2b, H, wub, H, ubuf, I, TOK, I, H, 1.f, nullptr, 0, stream);
        silu_mul_kernel<<<(unsigned)((nTI + 255) / 256), 256, 0, stream>>>(
            gbuf, ubuf, abuf, nTI);
        gemm_nt_f32(abuf, I, wdb, I, ybuf, H, TOK, H, I, 1.f, nullptr, 0, stream);
        moe_accum_kernel<<<(unsigned)((nTH + 255) / 256), 256, 0, stream>>>(
            ybuf, comb, e, out, nTH, H, E);
    }
}
